// Llama4TextExperts_37151467110421
// MI455X (gfx1250) — matrix operand order repaired and checked by reading, not compiled, not run on hardware
//
#include <hip/hip_runtime.h>
#include <hip/hip_bf16.h>
#include <stdint.h>

#define E_NUM  16
#define HID    2048
#define INTERD 4096
#define TOK    512

typedef __bf16 bf16;
typedef __attribute__((ext_vector_type(16))) bf16  v16bf;
typedef __attribute__((ext_vector_type(8)))  bf16  bf16x8;
typedef __attribute__((ext_vector_type(4)))  bf16  bf16x4;
typedef __attribute__((ext_vector_type(2)))  bf16  bf16x2;
typedef __attribute__((ext_vector_type(8)))  float v8f;
typedef __attribute__((ext_vector_type(4)))  float f32x4;

union AF { v16bf v; bf16x8 h[2]; };

__device__ __forceinline__ bf16x2 pack2(float a, float b) {
    bf16x2 r; r.x = (bf16)a; r.y = (bf16)b; return r;   // -> v_cvt_pk_bf16_f32
}

// ---------------------------------------------------------------------------
// Kernel 1: per-expert GEMM  hs(512x2048) x W1(2048x8192), fused SwiGLU,
// writes inter (E,512,4096) as bf16 to workspace.
// Block tile: 128 rows x 64 inter-cols (gate col j with up col j+4096).
// Double-buffered LDS, K-step 32 (bf16 WMMA depth).
// ---------------------------------------------------------------------------
#define LDA1 40
#define LDB1 40

struct Regs1 { f32x4 a[4]; f32x4 g[2]; f32x4 u[2]; };

__device__ __forceinline__ void load_regs1(Regs1& R, const float* A,
                                           const float* Bg, const float* Bu,
                                           int k0, int tid) {
    const int r  = tid >> 3;              // 0..31
    const int c4 = (tid & 7) * 4;         // 0..28
    #pragma unroll
    for (int p = 0; p < 4; ++p)
        R.a[p] = *(const f32x4*)(A + (size_t)(r + p * 32) * HID + k0 + c4);
    const int kp  = tid >> 4;             // 0..15 -> k pair (2kp, 2kp+1)
    const int bc4 = (tid & 15) * 4;       // 0..60
    const size_t ro0 = (size_t)(k0 + 2 * kp) * (2 * INTERD);
    const size_t ro1 = ro0 + (2 * INTERD);
    R.g[0] = *(const f32x4*)(Bg + ro0 + bc4);
    R.g[1] = *(const f32x4*)(Bg + ro1 + bc4);
    R.u[0] = *(const f32x4*)(Bu + ro0 + bc4);
    R.u[1] = *(const f32x4*)(Bu + ro1 + bc4);
}

__device__ __forceinline__ void store_lds1(const Regs1& R, bf16* sA,
                                           bf16* sBg, bf16* sBu, int tid) {
    const int r  = tid >> 3;
    const int c4 = (tid & 7) * 4;
    #pragma unroll
    for (int p = 0; p < 4; ++p) {
        bf16x4 v;
        v.x = (bf16)R.a[p].x; v.y = (bf16)R.a[p].y;
        v.z = (bf16)R.a[p].z; v.w = (bf16)R.a[p].w;
        *(bf16x4*)&sA[(r + p * 32) * LDA1 + c4] = v;
    }
    const int kp  = tid >> 4;
    const int bc4 = (tid & 15) * 4;
    #pragma unroll
    for (int c = 0; c < 4; ++c) {
        *(bf16x2*)&sBg[(bc4 + c) * LDB1 + 2 * kp] = pack2(R.g[0][c], R.g[1][c]);
        *(bf16x2*)&sBu[(bc4 + c) * LDB1 + 2 * kp] = pack2(R.u[0][c], R.u[1][c]);
    }
}

__global__ __launch_bounds__(256) void moe_gemm1_swiglu(
    const float* __restrict__ hs, const float* __restrict__ gup,
    bf16* __restrict__ inter)
{
    __shared__ __attribute__((aligned(16))) bf16 sA [2][128 * LDA1];
    __shared__ __attribute__((aligned(16))) bf16 sBg[2][ 64 * LDB1];
    __shared__ __attribute__((aligned(16))) bf16 sBu[2][ 64 * LDB1];

    const int e  = blockIdx.z;
    const int m0 = blockIdx.y * 128;
    const int n0 = blockIdx.x * 64;

    const float* A   = hs  + ((size_t)e * TOK + m0) * HID;
    const float* Bg  = gup + (size_t)e * HID * (2 * INTERD) + n0;
    const float* Bu  = Bg  + INTERD;
    bf16*        out = inter + ((size_t)e * TOK + m0) * (size_t)INTERD + n0;

    const int tid  = threadIdx.x;
    const int wave = tid >> 5;
    const int lane = tid & 31;
    const int ln   = lane & 15;
    const int hi   = lane >> 4;
    const int wm   = wave & 3;   // 32-row block within 128
    const int wn   = wave >> 2;  // 32-col block within 64

    v8f zero = {};
    v8f accG[2][2], accU[2][2];
    #pragma unroll
    for (int i = 0; i < 2; ++i)
        #pragma unroll
        for (int j = 0; j < 2; ++j) { accG[i][j] = zero; accU[i][j] = zero; }

    Regs1 R;
    load_regs1(R, A, Bg, Bu, 0, tid);
    store_lds1(R, sA[0], sBg[0], sBu[0], tid);

    const int NK = HID / 32;
    int buf = 0;
    for (int i = 0; i < NK; ++i) {
        if (i + 1 < NK) {
            load_regs1(R, A, Bg, Bu, (i + 1) * 32, tid);
            if (i + 2 < NK)
                __builtin_prefetch(Bg + (size_t)((i + 2) * 32) * (2 * INTERD) + ((tid & 15) * 4), 0, 0);
        }
        __syncthreads();

        AF a[2];
        #pragma unroll
        for (int mt = 0; mt < 2; ++mt) {
            const bf16* p = &sA[buf][(wm * 32 + mt * 16 + ln) * LDA1 + hi * 8];
            a[mt].h[0] = *(const bf16x8*)p;
            a[mt].h[1] = *(const bf16x8*)(p + 16);
        }
        AF bg[2], bu[2];
        #pragma unroll
        for (int nt = 0; nt < 2; ++nt) {
            const int col = wn * 32 + nt * 16 + ln;
            const bf16* pg = &sBg[buf][col * LDB1 + hi * 8];
            bg[nt].h[0] = *(const bf16x8*)pg;
            bg[nt].h[1] = *(const bf16x8*)(pg + 16);
            const bf16* pu = &sBu[buf][col * LDB1 + hi * 8];
            bu[nt].h[0] = *(const bf16x8*)pu;
            bu[nt].h[1] = *(const bf16x8*)(pu + 16);
        }
        #pragma unroll
        for (int mt = 0; mt < 2; ++mt)
            #pragma unroll
            for (int nt = 0; nt < 2; ++nt) {
                accG[mt][nt] = __builtin_amdgcn_wmma_f32_16x16x32_bf16(
                    false, a[mt].v, false, bg[nt].v, (short)0, accG[mt][nt], false, false);
                accU[mt][nt] = __builtin_amdgcn_wmma_f32_16x16x32_bf16(
                    false, a[mt].v, false, bu[nt].v, (short)0, accU[mt][nt], false, false);
            }

        if (i + 1 < NK) {
            store_lds1(R, sA[buf ^ 1], sBg[buf ^ 1], sBu[buf ^ 1], tid);
            buf ^= 1;
        }
    }

    // --- SwiGLU epilogue, store bf16 inter
    const int rbase = wm * 32 + hi * 8;
    #pragma unroll
    for (int mt = 0; mt < 2; ++mt)
        #pragma unroll
        for (int nt = 0; nt < 2; ++nt) {
            const int col = wn * 32 + nt * 16 + ln;
            #pragma unroll
            for (int r = 0; r < 8; ++r) {
                float g = accG[mt][nt][r];
                float u = accU[mt][nt][r];
                float s = u * g * __frcp_rn(1.0f + __expf(-g));
                out[(size_t)(rbase + mt * 16 + r) * INTERD + col] = (bf16)s;
            }
        }
}

// ---------------------------------------------------------------------------
// Kernel 2: per-expert GEMM  inter(512x4096, bf16) x W2(4096x2048) -> fp32.
// Block tile 128x128, double-buffered LDS, K-step 32.
// ---------------------------------------------------------------------------
#define LDA2 40
#define LDB2 40

struct Regs2 { bf16x8 a[2]; f32x4 b[2][2]; };

__device__ __forceinline__ void load_regs2(Regs2& R, const bf16* A,
                                           const float* B, int k0, int tid) {
    const int seg = tid & 3;              // 8-bf16 segment within row
    const int r0  = tid >> 2;             // 0..63
    #pragma unroll
    for (int p = 0; p < 2; ++p)
        R.a[p] = *(const bf16x8*)(A + (size_t)(r0 + p * 64) * INTERD + k0 + seg * 8);
    const int c4 = (tid & 31) * 4;        // 0..124
    const int kp = tid >> 5;              // 0..7
    #pragma unroll
    for (int p = 0; p < 2; ++p) {
        const int k = 2 * (kp + p * 8);   // k pair base
        R.b[p][0] = *(const f32x4*)(B + (size_t)(k0 + k) * HID + c4);
        R.b[p][1] = *(const f32x4*)(B + (size_t)(k0 + k + 1) * HID + c4);
    }
}

__device__ __forceinline__ void store_lds2(const Regs2& R, bf16* sA, bf16* sB,
                                           int tid) {
    const int seg = tid & 3;
    const int r0  = tid >> 2;
    #pragma unroll
    for (int p = 0; p < 2; ++p)
        *(bf16x8*)&sA[(r0 + p * 64) * LDA2 + seg * 8] = R.a[p];
    const int c4 = (tid & 31) * 4;
    const int kp = tid >> 5;
    #pragma unroll
    for (int p = 0; p < 2; ++p) {
        const int k = 2 * (kp + p * 8);
        #pragma unroll
        for (int c = 0; c < 4; ++c)
            *(bf16x2*)&sB[(c4 + c) * LDB2 + k] = pack2(R.b[p][0][c], R.b[p][1][c]);
    }
}

__global__ __launch_bounds__(256) void moe_gemm2(
    const bf16* __restrict__ inter, const float* __restrict__ dw,
    float* __restrict__ out)
{
    __shared__ __attribute__((aligned(16))) bf16 sA[2][128 * LDA2];
    __shared__ __attribute__((aligned(16))) bf16 sB[2][128 * LDB2];

    const int e  = blockIdx.z;
    const int m0 = blockIdx.y * 128;
    const int n0 = blockIdx.x * 128;

    const bf16*  A = inter + ((size_t)e * TOK + m0) * (size_t)INTERD;
    const float* B = dw + (size_t)e * INTERD * HID + n0;
    float*       O = out + ((size_t)e * TOK + m0) * (size_t)HID + n0;

    const int tid  = threadIdx.x;
    const int wave = tid >> 5;
    const int lane = tid & 31;
    const int ln   = lane & 15;
    const int hi   = lane >> 4;
    const int wm   = wave & 3;   // 32-row block within 128
    const int wn   = wave >> 2;  // 64-col block within 128

    v8f zero = {};
    v8f acc[2][4];
    #pragma unroll
    for (int i = 0; i < 2; ++i)
        #pragma unroll
        for (int j = 0; j < 4; ++j) acc[i][j] = zero;

    Regs2 R;
    load_regs2(R, A, B, 0, tid);
    store_lds2(R, sA[0], sB[0], tid);

    const int NK = INTERD / 32;
    int buf = 0;
    for (int i = 0; i < NK; ++i) {
        if (i + 1 < NK) {
            load_regs2(R, A, B, (i + 1) * 32, tid);
            if (i + 2 < NK)
                __builtin_prefetch(B + (size_t)((i + 2) * 32) * HID + ((tid & 31) * 4), 0, 0);
        }
        __syncthreads();

        AF a[2];
        #pragma unroll
        for (int mt = 0; mt < 2; ++mt) {
            const bf16* p = &sA[buf][(wm * 32 + mt * 16 + ln) * LDA2 + hi * 8];
            a[mt].h[0] = *(const bf16x8*)p;
            a[mt].h[1] = *(const bf16x8*)(p + 16);
        }
        AF b[4];
        #pragma unroll
        for (int nt = 0; nt < 4; ++nt) {
            const bf16* pb = &sB[buf][(wn * 64 + nt * 16 + ln) * LDB2 + hi * 8];
            b[nt].h[0] = *(const bf16x8*)pb;
            b[nt].h[1] = *(const bf16x8*)(pb + 16);
        }
        #pragma unroll
        for (int mt = 0; mt < 2; ++mt)
            #pragma unroll
            for (int nt = 0; nt < 4; ++nt)
                acc[mt][nt] = __builtin_amdgcn_wmma_f32_16x16x32_bf16(
                    false, a[mt].v, false, b[nt].v, (short)0, acc[mt][nt], false, false);

        if (i + 1 < NK) {
            store_lds2(R, sA[buf ^ 1], sB[buf ^ 1], tid);
            buf ^= 1;
        }
    }

    const int rbase = wm * 32 + hi * 8;
    #pragma unroll
    for (int mt = 0; mt < 2; ++mt)
        #pragma unroll
        for (int nt = 0; nt < 4; ++nt) {
            const int col = wn * 64 + nt * 16 + ln;
            #pragma unroll
            for (int r = 0; r < 8; ++r)
                O[(size_t)(rbase + mt * 16 + r) * HID + col] = acc[mt][nt][r];
        }
}

// ---------------------------------------------------------------------------
extern "C" void kernel_launch(void* const* d_in, const int* in_sizes, int n_in,
                              void* d_out, int out_size, void* d_ws, size_t ws_size,
                              hipStream_t stream) {
    (void)in_sizes; (void)n_in; (void)out_size; (void)ws_size;
    const float* hs  = (const float*)d_in[0];  // (E*T, H)
    const float* gup = (const float*)d_in[1];  // (E, H, 2I)
    const float* dwn = (const float*)d_in[2];  // (E, I, H)
    float*       out = (float*)d_out;          // (E*T, H)
    bf16*        itr = (bf16*)d_ws;            // (E, T, I) bf16 = 64 MiB

    dim3 blk(256, 1, 1);
    dim3 g1(INTERD / 64, TOK / 128, E_NUM);    // 64 x 4 x 16
    moe_gemm1_swiglu<<<g1, blk, 0, stream>>>(hs, gup, itr);

    dim3 g2(HID / 128, TOK / 128, E_NUM);      // 16 x 4 x 16
    moe_gemm2<<<g2, blk, 0, stream>>>(itr, dwn, out);
}
